// Block_33775622815825
// MI455X (gfx1250) — matrix operand order repaired and checked by reading, not compiled, not run on hardware
//
#include <hip/hip_runtime.h>
#include <hip/hip_bf16.h>
#include <hip/hip_fp16.h>

typedef __attribute__((ext_vector_type(16))) _Float16 v16h;
typedef __attribute__((ext_vector_type(8)))  float    v8f;

#define WMMA_F16(a, b, c) \
  __builtin_amdgcn_wmma_f32_16x16x32_f16(false, (a), false, (b), (short)0, (c), false, false)

// ---------------------------------------------------------------------------
// CDNA5 async global->LDS copy (ASYNCcnt-tracked), 16B per lane per issue.
// VDST = VGPR holding LDS byte address, VADDR = 64-bit global address.
// ---------------------------------------------------------------------------
__device__ __forceinline__ void async_copy_b128(void* lds, const void* g) {
  asm volatile("global_load_async_to_lds_b128 %0, %1, off"
               :: "v"((unsigned)(size_t)lds),
                  "v"((unsigned long long)(size_t)g)
               : "memory");
}

#if defined(__has_builtin)
#if __has_builtin(__builtin_amdgcn_s_wait_asynccnt)
#define WAIT_ASYNC0() __builtin_amdgcn_s_wait_asynccnt(0)
#endif
#endif
#ifndef WAIT_ASYNC0
#define WAIT_ASYNC0() asm volatile("s_wait_asynccnt 0x0" ::: "memory")
#endif

// ---------------------------------------------------------------------------
// WMMA fragment loaders (contiguous-K order, same in A and B; wave32)
// A 16x32 f16: lane<16 -> M=lane, K = {0..15}; lane>=16 -> K={16..31}
// ---------------------------------------------------------------------------
__device__ __forceinline__ v16h load_fragA(const _Float16* base, int ld) {
  const unsigned lane = threadIdx.x & 31u;
  const unsigned r = lane & 15u, hh = lane >> 4;
  const _Float16* p = base + (size_t)r * ld + hh * 16u;
  union { v16h v; uint4 q[2]; } u;
  u.q[0] = *(const uint4*)(p);
  u.q[1] = *(const uint4*)(p + 8);
  return u.v;
}

// B 32x16 f16 stored as Bt[N][K]: lane<16 -> N=lane, K={0..15}; lane>=16 -> K={16..31}
__device__ __forceinline__ v16h load_fragB(const _Float16* base, int ld) {
  const unsigned lane = threadIdx.x & 31u;
  const unsigned n = lane & 15u, hh = lane >> 4;
  const _Float16* p = base + (size_t)n * ld + hh * 16u;
  union { v16h v; uint4 q[2]; } u;
  u.q[0] = *(const uint4*)(p);
  u.q[1] = *(const uint4*)(p + 8);
  return u.v;
}

// ---------------------------------------------------------------------------
// Weight prep: fp32 -> f16, transposed to [N][K] for the B operand
// ---------------------------------------------------------------------------
__global__ void prep_headw(const float* __restrict__ w, _Float16* __restrict__ o,
                           int H, int C, int D) {
  size_t i = (size_t)blockIdx.x * blockDim.x + threadIdx.x;
  size_t tot = (size_t)H * C * D;
  if (i >= tot) return;
  int d = (int)(i % D);
  int c = (int)((i / D) % C);
  int h = (int)(i / ((size_t)C * D));
  o[((size_t)h * D + d) * C + c] = (_Float16)w[i];   // [H*D][C]
}

__global__ void prep_T(const float* __restrict__ w, _Float16* __restrict__ o,
                       int K, int N) {
  size_t i = (size_t)blockIdx.x * blockDim.x + threadIdx.x;
  size_t tot = (size_t)K * N;
  if (i >= tot) return;
  int n = (int)(i % N);
  int k = (int)(i / N);
  o[(size_t)n * K + k] = (_Float16)w[i];             // [N][K]
}

// ---------------------------------------------------------------------------
// LayerNorm: one block per token row (C = 1024), fp32 in, f16 out
// ---------------------------------------------------------------------------
__global__ __launch_bounds__(256) void layernorm_f16(
    const float* __restrict__ x, const float* __restrict__ g,
    const float* __restrict__ b, _Float16* __restrict__ out, int C) {
  const int row = blockIdx.x;
  const float* xr = x + (size_t)row * C;
  __shared__ float red[256];
  const int tid = threadIdx.x;

  float s = 0.f;
  for (int i = tid; i < C; i += 256) s += xr[i];
  red[tid] = s; __syncthreads();
  for (int st = 128; st > 0; st >>= 1) { if (tid < st) red[tid] += red[tid + st]; __syncthreads(); }
  const float mu = red[0] / (float)C;
  __syncthreads();

  float vs = 0.f;
  for (int i = tid; i < C; i += 256) { float d = xr[i] - mu; vs += d * d; }
  red[tid] = vs; __syncthreads();
  for (int st = 128; st > 0; st >>= 1) { if (tid < st) red[tid] += red[tid + st]; __syncthreads(); }
  const float rstd = rsqrtf(red[0] / (float)C + 1e-5f);

  _Float16* orow = out + (size_t)row * C;
  for (int i = tid; i < C; i += 256)
    orow[i] = (_Float16)((xr[i] - mu) * rstd * g[i] + b[i]);
}

// ---------------------------------------------------------------------------
// WMMA GEMM with double-buffered async-LDS pipeline:
//   out = epilogue( A[M][K] * Bt[N][K]^T )
// 256 threads = 8 waves; block tile 128(M) x 64(N); BK = 64.
// Per K-step per wave: 2 A-frags, 8 B-frags, 8 WMMAs, 1 barrier; async copies
// for the next tile are in flight during the WMMAs.
// ---------------------------------------------------------------------------
template <bool BIAS, bool RELU, bool RESID, bool OUTF16>
__global__ __launch_bounds__(256) void gemm_f16(
    const _Float16* __restrict__ A, const _Float16* __restrict__ Bt,
    const float* __restrict__ bias, const float* __restrict__ resid,
    void* __restrict__ outp, int M, int N, int K) {
  constexpr int LDA = 72;  // 64 + 8 pad (f16) -> 144B row stride, 16B aligned
  __shared__ __align__(16) _Float16 As[2][128 * LDA];
  __shared__ __align__(16) _Float16 Bs[2][64 * LDA];

  const int tid = threadIdx.x;
  const int w = tid >> 5;
  const unsigned lane = tid & 31u;
  const int n0 = blockIdx.x * 64;
  const int m0 = blockIdx.y * 128;

  // Stage one 128x64 A tile (1024 x 16B) + 64x64 Bt tile (512 x 16B):
  // 6 async instructions per wave -> ASYNCcnt += 6.
  auto stage = [&](int buf, int k0) {
    for (int i = tid; i < 1024; i += 256) {
      int r = i >> 3, c = i & 7;
      async_copy_b128(&As[buf][r * LDA + c * 8],
                      &A[(size_t)(m0 + r) * K + k0 + c * 8]);
    }
    for (int i = tid; i < 512; i += 256) {
      int r = i >> 3, c = i & 7;
      async_copy_b128(&Bs[buf][r * LDA + c * 8],
                      &Bt[(size_t)(n0 + r) * K + k0 + c * 8]);
    }
  };

  v8f acc[4] = {{}, {}, {}, {}};

  stage(0, 0);
  WAIT_ASYNC0();
  __syncthreads();

  int cur = 0;
  for (int k0 = 0; k0 < K; k0 += 64) {
    const bool more = (k0 + 64) < K;
    if (more) stage(cur ^ 1, k0 + 64);  // overlap next-tile HBM traffic with WMMAs

#pragma unroll
    for (int ks = 0; ks < 2; ++ks) {
      const v16h a = load_fragA(&As[cur][(w * 16) * LDA + ks * 32], LDA);
#pragma unroll
      for (int nt = 0; nt < 4; ++nt) {
        const v16h bfr = load_fragB(&Bs[cur][(nt * 16) * LDA + ks * 32], LDA);
        acc[nt] = WMMA_F16(a, bfr, acc[nt]);
      }
    }

    if (more) WAIT_ASYNC0();
    __syncthreads();
    cur ^= 1;
  }

  // epilogue: C layout -> lane<16: N=lane, M=j ; lane>=16: N=lane-16, M=8+j
  const unsigned coll = lane & 15u;
  const unsigned rj = (lane >> 4) * 8u;
#pragma unroll
  for (int nt = 0; nt < 4; ++nt) {
    const int coln = n0 + nt * 16 + (int)coll;
    const float bv = BIAS ? bias[coln] : 0.f;
#pragma unroll
    for (int j = 0; j < 8; ++j) {
      const size_t row = (size_t)m0 + w * 16 + rj + j;
      float vv = acc[nt][j] + bv;
      if (RELU) vv = fmaxf(vv, 0.f);
      if (RESID) vv += resid[row * N + coln];
      if (OUTF16) ((_Float16*)outp)[row * N + coln] = (_Float16)vv;
      else        ((float*)outp)[row * N + coln] = vv;
    }
  }
}

// ---------------------------------------------------------------------------
// Causal attention, one workgroup per (batch, head). T=128, Dh=64, scale=1/32.
// q/k/v/o: f16 [B*T][1024], head h occupies cols [h*64, h*64+64).
// ---------------------------------------------------------------------------
__global__ __launch_bounds__(256) void attention_f16(
    const _Float16* __restrict__ q, const _Float16* __restrict__ k,
    const _Float16* __restrict__ v, _Float16* __restrict__ o) {
  constexpr int T = 128, HD = 1024, Dh = 64, Hn = 16;
  constexpr int LQK = 72;   // 64 + 8 pad
  constexpr int LPV = 136;  // 128 + 8 pad
  // shared pool with overlay: [VT | Q | K], P overlays the Q/K region in phase 2
  __shared__ __align__(16) _Float16 smem[64 * LPV + 2 * 128 * LQK];
  _Float16* VTs = smem;                 // [64][LPV]
  _Float16* Qs  = smem + 64 * LPV;      // [128][LQK]
  _Float16* Ks  = Qs + 128 * LQK;       // [128][LQK]
  _Float16* Ps  = smem + 64 * LPV;      // [128][LPV], overlays Qs+Ks

  const int b = blockIdx.x / Hn, h = blockIdx.x % Hn;
  const size_t base = ((size_t)b * T) * HD + (size_t)h * Dh;
  const int tid = threadIdx.x;
  const int w = tid >> 5;
  const unsigned lane = tid & 31u;

  // stage Q, K (vectorized) and V transposed (scalar)
  for (int i = tid; i < 128 * 8; i += 256) {
    int r = i >> 3, c = i & 7;
    *(uint4*)&Qs[r * LQK + c * 8] = *(const uint4*)&q[base + (size_t)r * HD + c * 8];
    *(uint4*)&Ks[r * LQK + c * 8] = *(const uint4*)&k[base + (size_t)r * HD + c * 8];
  }
  for (int i = tid; i < 128 * 64; i += 256) {
    int t = i >> 6, d = i & 63;
    VTs[d * LPV + t] = v[base + (size_t)t * HD + d];
  }
  __syncthreads();

  // Phase 1: S = Q K^T  (wave w -> rows [w*16, w*16+16), all 128 cols)
  v16h aq[2];
  aq[0] = load_fragA(&Qs[(w * 16) * LQK + 0], LQK);
  aq[1] = load_fragA(&Qs[(w * 16) * LQK + 32], LQK);
  v8f s[8];
#pragma unroll
  for (int nt = 0; nt < 8; ++nt) s[nt] = (v8f){};
#pragma unroll
  for (int nt = 0; nt < 8; ++nt) {
#pragma unroll
    for (int kk = 0; kk < 2; ++kk) {
      const v16h bf = load_fragB(&Ks[(nt * 16) * LQK + kk * 32], LQK);
      s[nt] = WMMA_F16(aq[kk], bf, s[nt]);
    }
  }
  __syncthreads();  // Q/K reads done everywhere; safe to overlay with P

  // masked softmax per row; one row lives across 16 lanes of a half-wave
  const unsigned coll = lane & 15u;
  const unsigned rj = (lane >> 4) * 8u;
#pragma unroll
  for (int j = 0; j < 8; ++j) {
    const int row = w * 16 + (int)rj + j;
    float m = -1e30f;
#pragma unroll
    for (int nt = 0; nt < 8; ++nt) {
      const int col = nt * 16 + (int)coll;
      float vv = s[nt][j] * 0.03125f;          // C**-0.5 with C = 1024
      if (col > row) vv = -1e30f;              // causal mask
      s[nt][j] = vv;
      m = fmaxf(m, vv);
    }
    for (int off = 1; off < 16; off <<= 1) m = fmaxf(m, __shfl_xor(m, off, 16));
    float sum = 0.f;
#pragma unroll
    for (int nt = 0; nt < 8; ++nt) {
      const float e = __expf(s[nt][j] - m);
      s[nt][j] = e;
      sum += e;
    }
    for (int off = 1; off < 16; off <<= 1) sum += __shfl_xor(sum, off, 16);
    const float inv = 1.f / sum;
#pragma unroll
    for (int nt = 0; nt < 8; ++nt)
      Ps[row * LPV + nt * 16 + coll] = (_Float16)(s[nt][j] * inv);
  }
  __syncthreads();

  // Phase 2: O = P V  (K = 128 -> 4 steps; N = 64 -> 4 tiles)
  v8f o4[4] = {{}, {}, {}, {}};
#pragma unroll
  for (int kk = 0; kk < 4; ++kk) {
    const v16h ap = load_fragA(&Ps[(w * 16) * LPV + kk * 32], LPV);
#pragma unroll
    for (int nt = 0; nt < 4; ++nt) {
      const v16h bf = load_fragB(&VTs[(nt * 16) * LPV + kk * 32], LPV);
      o4[nt] = WMMA_F16(ap, bf, o4[nt]);
    }
  }
#pragma unroll
  for (int nt = 0; nt < 4; ++nt) {
#pragma unroll
    for (int j = 0; j < 8; ++j) {
      const int row = w * 16 + (int)rj + j;
      const int col = nt * 16 + (int)coll;
      o[base + (size_t)row * HD + col] = (_Float16)o4[nt][j];
    }
  }
}

// ---------------------------------------------------------------------------
// Host orchestration
// ---------------------------------------------------------------------------
extern "C" void kernel_launch(void* const* d_in, const int* in_sizes, int n_in,
                              void* d_out, int out_size, void* d_ws, size_t ws_size,
                              hipStream_t stream) {
  (void)in_sizes; (void)n_in; (void)out_size; (void)ws_size;

  const float* x      = (const float*)d_in[0];
  const float* wq     = (const float*)d_in[1];
  const float* wk     = (const float*)d_in[2];
  const float* wv     = (const float*)d_in[3];
  const float* w_proj = (const float*)d_in[4];
  const float* b_proj = (const float*)d_in[5];
  const float* w1     = (const float*)d_in[6];
  const float* b1     = (const float*)d_in[7];
  const float* w2     = (const float*)d_in[8];
  const float* b2     = (const float*)d_in[9];
  const float* ln1_g  = (const float*)d_in[10];
  const float* ln1_b  = (const float*)d_in[11];
  const float* ln2_g  = (const float*)d_in[12];
  const float* ln2_b  = (const float*)d_in[13];
  float* out = (float*)d_out;

  const int Bb = 128, T = 128, C = 1024, H = 16, D = 64, F = 4096;
  const size_t M = (size_t)Bb * T;  // 16384 tokens

  // workspace layout (f16 unless noted)
  _Float16* wqT = (_Float16*)d_ws;            // [C][C]  (Bt = [H*D][C])
  _Float16* wkT = wqT + (size_t)C * C;
  _Float16* wvT = wkT + (size_t)C * C;
  _Float16* wpT = wvT + (size_t)C * C;        // [C][C]
  _Float16* w1T = wpT + (size_t)C * C;        // [F][C]
  _Float16* w2T = w1T + (size_t)F * C;        // [C][F]
  _Float16* h1  = w2T + (size_t)C * F;        // [M][C]
  _Float16* qb  = h1 + M * C;                 // [M][C]
  _Float16* kb  = qb + M * C;
  _Float16* vb  = kb + M * C;
  _Float16* ac  = vb + M * C;                 // attention concat [M][C]
  _Float16* h2  = ac + M * C;                 // [M][C]
  _Float16* hid = h2 + M * C;                 // [M][F]
  float*    x1  = (float*)(hid + M * F);      // [M][C] fp32 residual stream

  const dim3 blk(256);

  // 1) weight prep (fp32 -> f16, [N][K])
  prep_headw<<<dim3((unsigned)(((size_t)H * C * D + 255) / 256)), blk, 0, stream>>>(wq, wqT, H, C, D);
  prep_headw<<<dim3((unsigned)(((size_t)H * C * D + 255) / 256)), blk, 0, stream>>>(wk, wkT, H, C, D);
  prep_headw<<<dim3((unsigned)(((size_t)H * C * D + 255) / 256)), blk, 0, stream>>>(wv, wvT, H, C, D);
  prep_T<<<dim3((unsigned)(((size_t)C * C + 255) / 256)), blk, 0, stream>>>(w_proj, wpT, C, C);
  prep_T<<<dim3((unsigned)(((size_t)C * F + 255) / 256)), blk, 0, stream>>>(w1, w1T, C, F);
  prep_T<<<dim3((unsigned)(((size_t)F * C + 255) / 256)), blk, 0, stream>>>(w2, w2T, F, C);

  // 2) LN1
  layernorm_f16<<<dim3((unsigned)M), blk, 0, stream>>>(x, ln1_g, ln1_b, h1, C);

  // 3) QKV projections (WMMA GEMMs, f16 out)
  gemm_f16<false, false, false, true><<<dim3(C / 64, (unsigned)(M / 128)), blk, 0, stream>>>(
      h1, wqT, nullptr, nullptr, qb, (int)M, C, C);
  gemm_f16<false, false, false, true><<<dim3(C / 64, (unsigned)(M / 128)), blk, 0, stream>>>(
      h1, wkT, nullptr, nullptr, kb, (int)M, C, C);
  gemm_f16<false, false, false, true><<<dim3(C / 64, (unsigned)(M / 128)), blk, 0, stream>>>(
      h1, wvT, nullptr, nullptr, vb, (int)M, C, C);

  // 4) causal attention per (b, h)
  attention_f16<<<dim3(Bb * H), blk, 0, stream>>>(qb, kb, vb, ac);

  // 5) out-proj + bias + residual(x) -> x1 (fp32)
  gemm_f16<true, false, true, false><<<dim3(C / 64, (unsigned)(M / 128)), blk, 0, stream>>>(
      ac, wpT, b_proj, x, x1, (int)M, C, C);

  // 6) LN2
  layernorm_f16<<<dim3((unsigned)M), blk, 0, stream>>>(x1, ln2_g, ln2_b, h2, C);

  // 7) MLP up + ReLU (f16 out)
  gemm_f16<true, true, false, true><<<dim3(F / 64, (unsigned)(M / 128)), blk, 0, stream>>>(
      h2, w1T, b1, nullptr, hid, (int)M, F, C);

  // 8) MLP down + bias + residual(x1) -> d_out (fp32)
  gemm_f16<true, false, true, false><<<dim3(C / 64, (unsigned)(M / 128)), blk, 0, stream>>>(
      hid, w2T, b2, x1, out, (int)M, C, F);
}
